// Model_43628277793444
// MI455X (gfx1250) — matrix operand order repaired and checked by reading, not compiled, not run on hardware
//
#include <hip/hip_runtime.h>
#include <hip/hip_bf16.h>

// ---------------------------------------------------------------------------
// GPT-style forward for MI455X (gfx1250, wave32).
// bf16 operands everywhere, v_wmma_f32_16x16x32_bf16 compute, Tensor Data
// Mover (tensor_load_to_lds + TENSORcnt) double-buffered LDS staging.
// ---------------------------------------------------------------------------

#define BDIM   1024   // model dim D
#define FDIM   4096   // FFN dim
#define SDIM   2048   // sequence length
#define BATCH  2
#define VDIM   32000
#define NLAYER 6

typedef __attribute__((ext_vector_type(16))) __bf16 v16bf;
typedef __attribute__((ext_vector_type(8)))  __bf16 v8bf;
typedef __attribute__((ext_vector_type(4)))  __bf16 v4bf;
typedef __attribute__((ext_vector_type(8)))  float  v8f;
typedef __attribute__((ext_vector_type(4)))  unsigned int v4u;
typedef __attribute__((ext_vector_type(8)))  int v8i;
typedef __attribute__((ext_vector_type(4)))  int v4i;

#define LDSA   72                  // padded LDS row stride (64 data + 8 pad bf16)
#define TILE_E (128 * LDSA)        // bf16 elements per LDS tile buffer
#define TILE_B (TILE_E * 2)        // bytes per LDS tile buffer

#ifndef __has_builtin
#define __has_builtin(x) 0
#endif
#if __has_builtin(__builtin_amdgcn_tensor_load_to_lds)
#define HAVE_TDM 1
#else
#define HAVE_TDM 0
#endif

#if HAVE_TDM
// 2D tile load via Tensor Data Mover: tile_dim0=64 (K, contiguous bf16),
// tile_dim1=128 rows, row stride = strideElems, LDS padding 4 DWORDs after
// every 32 DWORDs -> LDSA=72 row stride in LDS. Descriptor per ISA ch.8.
// This toolchain exposes the 6-arg builtin:
//   (uint32x4 g0, int32x8 g1, int32x4 g2, int32x4 g3, int32x8 extra, i32 cpol)
__device__ __forceinline__ void tdm_tile_load(unsigned ldsByteOff,
                                              const __bf16* gptr,
                                              int strideElems)
{
    const unsigned long long ga = (unsigned long long)(size_t)gptr;
    const v4u g0 = {
        1u,                                                  // count=1, user mode
        ldsByteOff,                                          // lds_addr (bytes)
        (unsigned)(ga & 0xFFFFFFFFu),                        // global_addr[31:0]
        (unsigned)((ga >> 32) & 0x01FFFFFFu) | (2u << 30)    // global_addr[56:32] | type=2
    };
    const unsigned td = 1u << 30;                            // huge tensor dims (tiles in-bounds)
    const v8i g1 = {
        (int)((1u << 16)        // data_size = 2 bytes
            | (1u << 20)        // pad_enable
            | (4u << 22)        // pad_interval: 32 DWORDs (=64 bf16)
            | (3u << 25)),      // pad_amount  : 4 DWORDs  (=8 bf16)
        (int)((td & 0xFFFFu) << 16),                            // tensor_dim0[15:0]
        (int)(((td >> 16) & 0xFFFFu) | ((td & 0xFFFFu) << 16)), // td0[31:16] | td1[15:0]
        (int)(((td >> 16) & 0xFFFFu) | (64u << 16)),            // td1[31:16] | tile_dim0=64
        (int)(128u),                                            // tile_dim1=128, tile_dim2=0
        (int)(unsigned)strideElems,                             // tensor_dim0_stride[31:0]
        0,                                                      // stride0[47:32] | stride1[15:0]
        0                                                       // stride1[47:16]
    };
    const v4i gz4 = {0, 0, 0, 0};
    const v8i gz8 = {0, 0, 0, 0, 0, 0, 0, 0};
    __builtin_amdgcn_tensor_load_to_lds(g0, g1, gz4, gz4, gz8, 0);
}

__device__ __forceinline__ void tdm_wait(int n)
{
#if __has_builtin(__builtin_amdgcn_s_wait_tensorcnt)
    if (n == 0) __builtin_amdgcn_s_wait_tensorcnt(0);
    else        __builtin_amdgcn_s_wait_tensorcnt(2);
#else
    if (n == 0) asm volatile("s_wait_tensorcnt 0x0" ::: "memory");
    else        asm volatile("s_wait_tensorcnt 0x2" ::: "memory");
#endif
}
#endif // HAVE_TDM

// ---------------------------------------------------------------------------
// Tiled bf16 WMMA GEMM:  D[M,N] = A[M,K] @ W[N,K]^T (+bias)(+relu)(+residual)
// EPI bitmask: 1=+bias[col], 2=relu, 4=+R[row,col].  TOUT = float or __bf16.
// Block tile 128x128, K tile 64, 256 threads = 8 waves (2x4 grid), each wave
// 4x2 fragments of 16x16x32.  LDS double-buffered: [A0][A1][B0][B1] = 72KB.
// All M,N multiples of 128, K multiples of 128 (true for every call here).
// ---------------------------------------------------------------------------
template<int EPI, typename TOUT>
__global__ __launch_bounds__(256) void gemm_kernel(
    const __bf16* __restrict__ A, int lda, long sA,
    const __bf16* __restrict__ W, int ldw, long sW,
    const float* __restrict__ bias,
    const float* __restrict__ R, int ldr, long sR,
    TOUT* __restrict__ Dst, int ldd, long sD,
    int K)
{
    __shared__ __bf16 smem[4 * TILE_E];

    const int tid  = threadIdx.x;
    const int lane = tid & 31;
    const int lrow = lane & 15;
    const int lhi  = lane >> 4;
    const int w    = tid >> 5;
    const int wm   = w >> 2;       // 0..1
    const int wn   = w & 3;        // 0..3

    const int m0 = blockIdx.y * 128;
    const int n0 = blockIdx.x * 128;

    const __bf16* Ab = A + (size_t)blockIdx.z * sA + (size_t)m0 * lda;
    const __bf16* Wb = W + (size_t)blockIdx.z * sW + (size_t)n0 * ldw;
    TOUT*         Db = Dst + (size_t)blockIdx.z * sD;

    v8f acc[4][2];
    const v8f vzero = {0.f, 0.f, 0.f, 0.f, 0.f, 0.f, 0.f, 0.f};
#pragma unroll
    for (int i = 0; i < 4; ++i)
#pragma unroll
        for (int j = 0; j < 2; ++j) acc[i][j] = vzero;

    const int niter = K >> 6;
#if HAVE_TDM
    const unsigned smemOff = (unsigned)(size_t)(void*)smem;  // LDS byte offset
    if (tid < 32) {                                          // wave 0 drives the TDM
        tdm_tile_load(smemOff + 0 * TILE_B, Ab, lda);
        tdm_tile_load(smemOff + 2 * TILE_B, Wb, ldw);
    }
#endif

    for (int it = 0; it < niter; ++it) {
        const int par = it & 1;
        __bf16* As = smem + par * TILE_E;
        __bf16* Bs = smem + (2 + par) * TILE_E;

#if HAVE_TDM
        if (tid < 32) {
            if (it + 1 < niter) {       // prefetch next tile pair, keep 1 pair in flight
                const int p2  = (it + 1) & 1;
                const int kt2 = (it + 1) << 6;
                tdm_tile_load(smemOff + p2 * TILE_B, Ab + kt2, lda);
                tdm_tile_load(smemOff + (2 + p2) * TILE_B, Wb + kt2, ldw);
                tdm_wait(2);
            } else {
                tdm_wait(0);
            }
        }
        __syncthreads();
        asm volatile("" ::: "memory");  // TDM wrote LDS behind the compiler's back
#else
        // cooperative fallback: 128x64 bf16 tile, 8 bf16 per thread per load
        const int kt = it << 6;
#pragma unroll
        for (int i = 0; i < 4; ++i) {
            int idx = tid + i * 256;          // 0..1023
            int r   = idx >> 3;               // row 0..127
            int c8  = (idx & 7) << 3;         // 0,8,..,56
            *(v8bf*)(&As[r * LDSA + c8]) = *(const v8bf*)(Ab + (size_t)r * lda + kt + c8);
            *(v8bf*)(&Bs[r * LDSA + c8]) = *(const v8bf*)(Wb + (size_t)r * ldw + kt + c8);
        }
        __syncthreads();
#endif

#pragma unroll
        for (int kk = 0; kk < 64; kk += 32) {
            // A frag 16x32: lanes 0-15 hold M=lane, K 0..7 & 16..23; lanes 16-31 K 8..15 & 24..31
            v16bf afrag[4];
#pragma unroll
            for (int mi = 0; mi < 4; ++mi) {
                const __bf16* p = &As[(wm * 64 + mi * 16 + lrow) * LDSA + kk + lhi * 8];
                v8bf lo = *(const v8bf*)p;
                v8bf hi = *(const v8bf*)(p + 16);
                afrag[mi] = __builtin_shufflevector(lo, hi,
                    0,1,2,3,4,5,6,7,8,9,10,11,12,13,14,15);
            }
            // B frag 32x16: lanes 0-15 N=lane hold K 0..7 & 16..23; lanes 16-31 K 8..15 & 24..31
            v16bf bfrag[2];
#pragma unroll
            for (int ni = 0; ni < 2; ++ni) {
                const __bf16* p = &Bs[(wn * 32 + ni * 16 + lrow) * LDSA + kk + lhi * 8];
                v8bf lo = *(const v8bf*)p;
                v8bf hi = *(const v8bf*)(p + 16);
                bfrag[ni] = __builtin_shufflevector(lo, hi,
                    0,1,2,3,4,5,6,7,8,9,10,11,12,13,14,15);
            }
#pragma unroll
            for (int mi = 0; mi < 4; ++mi)
#pragma unroll
                for (int ni = 0; ni < 2; ++ni)
                    acc[mi][ni] = __builtin_amdgcn_wmma_f32_16x16x32_bf16(
                        false, afrag[mi], false, bfrag[ni],
                        (short)0, acc[mi][ni], false, false);
        }
        __syncthreads();
    }

    // epilogue: D frag VGPR r -> row r (+8 for lanes 16-31), col = lane&15
    const float* Rb = (R != nullptr) ? (R + (size_t)blockIdx.z * sR) : nullptr;
#pragma unroll
    for (int mi = 0; mi < 4; ++mi) {
#pragma unroll
        for (int ni = 0; ni < 2; ++ni) {
            const int col = n0 + wn * 32 + ni * 16 + lrow;
#pragma unroll
            for (int r = 0; r < 8; ++r) {
                const int row = m0 + wm * 64 + mi * 16 + lhi * 8 + r;
                float v = acc[mi][ni][r];
                if (EPI & 1) v += bias[col];
                if (EPI & 2) v = fmaxf(v, 0.0f);
                if (EPI & 4) v += Rb[(size_t)row * ldr + col];
                Db[(size_t)row * ldd + col] = (TOUT)v;
            }
        }
    }
}

// ---------------------------------------------------------------------------
// Embedding: h[b,s,:] = embed_table[tok[b,s],:] + pos_table[s,:]  (f32)
// ---------------------------------------------------------------------------
__global__ __launch_bounds__(256) void embed_kernel(
    const int* __restrict__ tok, const float* __restrict__ emb,
    const float* __restrict__ pos, float* __restrict__ h)
{
    const int row = blockIdx.x;
    const int s   = row & (SDIM - 1);
    const int t   = tok[row];
    for (int d = threadIdx.x; d < BDIM; d += 256)
        h[(size_t)row * BDIM + d] = emb[(size_t)t * BDIM + d] + pos[(size_t)s * BDIM + d];
}

// ---------------------------------------------------------------------------
// f32 -> bf16 conversion (n multiple of 4)
// ---------------------------------------------------------------------------
__global__ __launch_bounds__(256) void cvt_kernel(
    const float* __restrict__ s, __bf16* __restrict__ d, long n)
{
    const long i = ((long)blockIdx.x * 256 + threadIdx.x) * 4;
    if (i < n) {
        const float4 v = *(const float4*)(s + i);
        v4bf p = {(__bf16)v.x, (__bf16)v.y, (__bf16)v.z, (__bf16)v.w};
        *(v4bf*)(d + i) = p;
    }
}

// ---------------------------------------------------------------------------
// LayerNorm over D=1024, bf16 output (feeds GEMM only)
// ---------------------------------------------------------------------------
__global__ __launch_bounds__(256) void ln_kernel(
    const float* __restrict__ x, const float* __restrict__ w,
    const float* __restrict__ b, __bf16* __restrict__ o)
{
    const int row = blockIdx.x;
    const float* xr = x + (size_t)row * BDIM;
    __shared__ float red[256];
    const int tid = threadIdx.x;

    float s = 0.f;
    for (int d = tid; d < BDIM; d += 256) s += xr[d];
    red[tid] = s; __syncthreads();
    for (int off = 128; off > 0; off >>= 1) {
        if (tid < off) red[tid] += red[tid + off];
        __syncthreads();
    }
    const float mu = red[0] * (1.0f / BDIM);
    __syncthreads();

    float vs = 0.f;
    for (int d = tid; d < BDIM; d += 256) { float t = xr[d] - mu; vs += t * t; }
    red[tid] = vs; __syncthreads();
    for (int off = 128; off > 0; off >>= 1) {
        if (tid < off) red[tid] += red[tid + off];
        __syncthreads();
    }
    const float rinv = rsqrtf(red[0] * (1.0f / BDIM) + 1e-5f);

    for (int d = tid; d < BDIM; d += 256)
        o[(size_t)row * BDIM + d] = (__bf16)((xr[d] - mu) * rinv * w[d] + b[d]);
}

// ---------------------------------------------------------------------------
// Causal softmax: f32 scores in, bf16 probabilities out. scale = 1/sqrt(1024)
// ---------------------------------------------------------------------------
__global__ __launch_bounds__(256) void softmax_kernel(
    float* __restrict__ scores, __bf16* __restrict__ probs)
{
    const int rid = blockIdx.x;                 // b*S + s
    const int s   = rid & (SDIM - 1);
    float*   row = scores + (size_t)rid * SDIM;
    __bf16*  brow = probs + (size_t)rid * SDIM;
    __shared__ float red[256];
    const int tid = threadIdx.x;
    const float inv = 1.0f / 32.0f;

    float m = -3.4e38f;
    for (int t = tid; t < SDIM; t += 256)
        if (t <= s) m = fmaxf(m, row[t] * inv);
    red[tid] = m; __syncthreads();
    for (int off = 128; off > 0; off >>= 1) {
        if (tid < off) red[tid] = fmaxf(red[tid], red[tid + off]);
        __syncthreads();
    }
    m = red[0];
    __syncthreads();

    float sum = 0.f;
    for (int t = tid; t < SDIM; t += 256) {
        float e = (t <= s) ? __expf(row[t] * inv - m) : 0.f;
        row[t] = e;
        sum += e;
    }
    red[tid] = sum; __syncthreads();
    for (int off = 128; off > 0; off >>= 1) {
        if (tid < off) red[tid] += red[tid + off];
        __syncthreads();
    }
    const float rs = 1.0f / red[0];
    for (int t = tid; t < SDIM; t += 256) brow[t] = (__bf16)(row[t] * rs);
}

// ---------------------------------------------------------------------------
// vT[b][d][t] = qkv_bf16[b*S+t][2D+d]  (32x32 LDS tile transpose)
// ---------------------------------------------------------------------------
__global__ __launch_bounds__(256) void transpose_v_kernel(
    const __bf16* __restrict__ bqkv, __bf16* __restrict__ vT)
{
    __shared__ __bf16 tile[32][33];
    const int b  = blockIdx.z;
    const int t0 = blockIdx.x * 32;
    const int d0 = blockIdx.y * 32;
    const int tx = threadIdx.x & 31;
    const int ty = threadIdx.x >> 5;            // 0..7
#pragma unroll
    for (int i = 0; i < 32; i += 8)
        tile[ty + i][tx] =
            bqkv[(size_t)(b * SDIM + t0 + ty + i) * (3 * BDIM) + 2 * BDIM + d0 + tx];
    __syncthreads();
#pragma unroll
    for (int i = 0; i < 32; i += 8)
        vT[((size_t)b * BDIM + d0 + ty + i) * SDIM + t0 + tx] = tile[tx][ty + i];
}

// ---------------------------------------------------------------------------
// Host orchestration (all on `stream`; graph-capture safe)
// ---------------------------------------------------------------------------
extern "C" void kernel_launch(void* const* d_in, const int* in_sizes, int n_in,
                              void* d_out, int out_size, void* d_ws, size_t ws_size,
                              hipStream_t stream)
{
    const int*   tokens = (const int*)  d_in[0];
    const float* emb    = (const float*)d_in[1];
    const float* pos    = (const float*)d_in[2];
    const float* ln1w   = (const float*)d_in[3];
    const float* ln1b   = (const float*)d_in[4];
    const float* qkvw   = (const float*)d_in[5];
    const float* qkvb   = (const float*)d_in[6];
    const float* ln2w   = (const float*)d_in[7];
    const float* ln2b   = (const float*)d_in[8];
    const float* ffaw   = (const float*)d_in[9];
    const float* ffab   = (const float*)d_in[10];
    const float* ffbw   = (const float*)d_in[11];
    const float* ffbb   = (const float*)d_in[12];
    const float* outw   = (const float*)d_in[13];
    float* out = (float*)d_out;

    const int BS = BATCH * SDIM;                       // 4096 rows

    // workspace carve (~175 MB); scores(f32) and bff(bf16) share one region
    char* p = (char*)d_ws;
    auto alloc = [&](size_t bytes) { char* q = p; p += (bytes + 255) & ~(size_t)255; return q; };
    float*  h      = (float*) alloc((size_t)BS * BDIM * 4);            // 16.8 MB
    float*  scores = (float*) alloc((size_t)BATCH * SDIM * SDIM * 4);  // 33.6 MB (alias bff)
    __bf16* bff    = (__bf16*)scores;                                  // [4096,4096] bf16
    __bf16* bA     = (__bf16*)alloc((size_t)BS * BDIM * 2);            //  8.4 MB
    __bf16* bqkv   = (__bf16*)alloc((size_t)BS * 3 * BDIM * 2);        // 25.2 MB
    __bf16* bvT    = (__bf16*)alloc((size_t)BATCH * BDIM * SDIM * 2);  //  8.4 MB
    __bf16* battn  = (__bf16*)alloc((size_t)BATCH * SDIM * SDIM * 2);  // 16.8 MB
    __bf16* bW     = (__bf16*)alloc((size_t)VDIM * BDIM * 2);          // 65.5 MB

    const long sQ  = (long)SDIM * 3 * BDIM;            // per-batch stride in bqkv
    const long sSS = (long)SDIM * SDIM;                // per-batch stride in scores/battn
    const long sVT = (long)BDIM * SDIM;                // per-batch stride in bvT
    const long sSD = (long)SDIM * BDIM;                // per-batch stride in h

    auto cvt = [&](const float* src, __bf16* dst, long n) {
        cvt_kernel<<<dim3((unsigned)((n / 4 + 255) / 256)), 256, 0, stream>>>(src, dst, n);
    };

    embed_kernel<<<dim3(BS), 256, 0, stream>>>(tokens, emb, pos, h);

    for (int l = 0; l < NLAYER; ++l) {
        // bA = bf16(LN1(h))
        ln_kernel<<<dim3(BS), 256, 0, stream>>>(h, ln1w + (size_t)l * BDIM,
                                                ln1b + (size_t)l * BDIM, bA);
        // bqkv = bf16(bA @ qkv_w^T + qkv_b)   [4096 x 3072]
        cvt(qkvw + (size_t)l * 3 * BDIM * BDIM, bW, (long)3 * BDIM * BDIM);
        gemm_kernel<1, __bf16><<<dim3(3 * BDIM / 128, BS / 128, 1), 256, 0, stream>>>(
            bA, BDIM, 0, bW, BDIM, 0, qkvb + (size_t)l * 3 * BDIM,
            nullptr, 0, 0, bqkv, 3 * BDIM, 0, BDIM);
        // scores = q @ k^T per batch          [S x S] f32
        gemm_kernel<0, float><<<dim3(SDIM / 128, SDIM / 128, BATCH), 256, 0, stream>>>(
            bqkv, 3 * BDIM, sQ, bqkv + BDIM, 3 * BDIM, sQ,
            nullptr, nullptr, 0, 0, scores, SDIM, sSS, BDIM);
        // battn = bf16(causal softmax(scores / 32))
        softmax_kernel<<<dim3(BS), 256, 0, stream>>>(scores, battn);
        // bvT[b][d][t] = v
        transpose_v_kernel<<<dim3(SDIM / 32, BDIM / 32, BATCH), 256, 0, stream>>>(bqkv, bvT);
        // h = h + battn @ vT^T per batch      [S x D] f32
        gemm_kernel<4, float><<<dim3(BDIM / 128, SDIM / 128, BATCH), 256, 0, stream>>>(
            battn, SDIM, sSS, bvT, SDIM, sVT,
            nullptr, h, BDIM, sSD, h, BDIM, sSD, SDIM);
        // bA = bf16(LN2(h))
        ln_kernel<<<dim3(BS), 256, 0, stream>>>(h, ln2w + (size_t)l * BDIM,
                                                ln2b + (size_t)l * BDIM, bA);
        // bff = bf16(relu(bA @ ffa_w^T + ffa_b))   [4096 x 4096]
        cvt(ffaw + (size_t)l * FDIM * BDIM, bW, (long)FDIM * BDIM);
        gemm_kernel<3, __bf16><<<dim3(FDIM / 128, BS / 128, 1), 256, 0, stream>>>(
            bA, BDIM, 0, bW, BDIM, 0, ffab + (size_t)l * FDIM,
            nullptr, 0, 0, bff, FDIM, 0, BDIM);
        // h = h + bff @ ffb_w^T + ffb_b            [4096 x 1024]
        cvt(ffbw + (size_t)l * BDIM * FDIM, bW, (long)BDIM * FDIM);
        gemm_kernel<5, float><<<dim3(BDIM / 128, BS / 128, 1), 256, 0, stream>>>(
            bff, FDIM, 0, bW, FDIM, 0, ffbb + (size_t)l * BDIM,
            h, BDIM, 0, h, BDIM, 0, FDIM);
    }

    // logits = h @ out_w^T    [4096 x 32000] f32 -> d_out
    cvt(h, bA, (long)BS * BDIM);  // reuse bA as bf16(h)
    cvt(outw, bW, (long)VDIM * BDIM);
    gemm_kernel<0, float><<<dim3(VDIM / 128, BS / 128, 1), 256, 0, stream>>>(
        bA, BDIM, 0, bW, BDIM, 0,
        nullptr, nullptr, 0, 0, out, VDIM, 0, BDIM);
}
